// EurNetBlock_78262894068122
// MI455X (gfx1250) — compile-verified, matrix operand order checked by reading, not run on hardware
//
#include <hip/hip_runtime.h>
#include <hip/hip_bf16.h>
#include <math.h>

// ---------------- problem constants ----------------
constexpr int Bx  = 16;
constexpr int Lx  = 3136;
constexpr int Cx  = 96;
constexpr int Rr  = 3;
constexpr int RTt = 5;
constexpr int Ee  = 131072;
constexpr int Hh  = 56;
constexpr int Ww  = 56;
constexpr int FFN = 384;
constexpr size_t M = (size_t)Bx * Lx;     // 50176 tokens

typedef __attribute__((ext_vector_type(16))) _Float16 v16h;
typedef __attribute__((ext_vector_type(8)))  _Float16 v8h;
typedef __attribute__((ext_vector_type(8)))  float    v8f;

// ---------------- helpers ----------------
__device__ __forceinline__ float gelu_exact(float x) {
  return 0.5f * x * (1.0f + erff(x * 0.70710678118654752440f));
}

// Load one 16x32 f16 fragment (A or B^T), row-major, ld = K.
// ISA 7.12.2: lane l holds row (l&15); K-halves: [kb..kb+7] and [kb+16..kb+23], kb = (l>>4)*8.
__device__ __forceinline__ v16h load_frag(const _Float16* __restrict__ base,
                                          int ld, int row0, int k0, int lane) {
  int r  = row0 + (lane & 15);
  int kb = k0 + ((lane >> 4) << 3);
  const _Float16* p = base + (size_t)r * ld + kb;
  v8h lo = *(const v8h*)(p);
  v8h hi = *(const v8h*)(p + 16);
  v16h out;
#pragma unroll
  for (int i = 0; i < 8; ++i) { out[i] = lo[i]; out[i + 8] = hi[i]; }
  return out;
}

// ---------------- generic WMMA GEMM: OUT[M,N] = act(A[M,K] @ BT[N,K]^T + bias) (+resid) ----------
template <bool GELU, bool RESID, bool STORE32, bool STORE16>
__global__ __launch_bounds__(256) void wmma_gemm(const _Float16* __restrict__ A,
                                                 const _Float16* __restrict__ BT,
                                                 const float* __restrict__ bias,
                                                 const float* __restrict__ resid,
                                                 float* __restrict__ out32,
                                                 _Float16* __restrict__ out16,
                                                 int Mtiles, int Ntiles, int K) {
  int wave = (blockIdx.x << 3) + (threadIdx.x >> 5);
  if (wave >= Mtiles * Ntiles) return;      // whole-wave uniform exit
  int tm = wave / Ntiles;
  int tn = wave - tm * Ntiles;
  int lane = threadIdx.x & 31;
  int N = Ntiles << 4;

  v8f acc = {};
  for (int k0 = 0; k0 < K; k0 += 32) {
    v16h a = load_frag(A,  K, tm << 4, k0, lane);
    v16h b = load_frag(BT, K, tn << 4, k0, lane);
    acc = __builtin_amdgcn_wmma_f32_16x16x32_f16(
        false, a, false, b, (short)0, acc, false, false);
  }

  int n = (tn << 4) + (lane & 15);
  float bv = bias[n];
  int mbase = (tm << 4) + ((lane >> 4) << 3);   // D VGPR i -> row i (lanes 0-15) / i+8 (lanes 16-31)
#pragma unroll
  for (int i = 0; i < 8; ++i) {
    float x = acc[i] + bv;
    if (GELU) x = gelu_exact(x);
    size_t idx = (size_t)(mbase + i) * N + n;
    if (RESID) x += resid[idx];
    if (STORE32) out32[idx] = x;
    if (STORE16) out16[idx] = (_Float16)x;
  }
}

// ---------------- layernorm: one wave per token, 3 channels per lane ----------------
__global__ __launch_bounds__(256) void ln_kernel(const float* __restrict__ x,
                                                 const float* __restrict__ g,
                                                 const float* __restrict__ b,
                                                 _Float16* __restrict__ out16) {
  int token = (blockIdx.x << 3) + (threadIdx.x >> 5);
  if (token >= (int)M) return;
  int lane = threadIdx.x & 31;
  const float* xp = x + (size_t)token * Cx;
  float v0 = xp[lane], v1 = xp[lane + 32], v2 = xp[lane + 64];
  float s  = v0 + v1 + v2;
  float sq = v0 * v0 + v1 * v1 + v2 * v2;
#pragma unroll
  for (int m = 16; m > 0; m >>= 1) {
    s  += __shfl_xor(s,  m, 32);
    sq += __shfl_xor(sq, m, 32);
  }
  float mean = s * (1.0f / Cx);
  float var  = sq * (1.0f / Cx) - mean * mean;
  float inv  = rsqrtf(var + 1e-5f);
  _Float16* op = out16 + (size_t)token * Cx;
  op[lane]      = (_Float16)((v0 - mean) * inv * g[lane]      + b[lane]);
  op[lane + 32] = (_Float16)((v1 - mean) * inv * g[lane + 32] + b[lane + 32]);
  op[lane + 64] = (_Float16)((v2 - mean) * inv * g[lane + 64] + b[lane + 64]);
}

// ---------------- weight convert f32 [K,N] -> f16 [N,K] ----------------
__global__ void transpose_f16_kernel(const float* __restrict__ w, _Float16* __restrict__ o,
                                     int K, int N) {
  int idx = blockIdx.x * blockDim.x + threadIdx.x;
  if (idx >= K * N) return;
  int n = idx / K;
  int k = idx - n * K;
  o[idx] = (_Float16)w[(size_t)k * N + n];
}

// ---------------- edge segment counts (batch independent) ----------------
__global__ void count_kernel(const int* __restrict__ ei, const int* __restrict__ et,
                             float* __restrict__ cnt) {
  int e = blockIdx.x * blockDim.x + threadIdx.x;
  if (e >= Ee) return;
  int dst = ei[Ee + e];
  int r   = et[e];
  atomicAdd(&cnt[dst * Rr + r], 1.0f);
}

// ---------------- edge scatter-add: agg[b, dst*R+r, c] += v[b, src, c] ----------------
__global__ __launch_bounds__(256) void scatter_kernel(const float* __restrict__ v,
                                                      const int* __restrict__ ei,
                                                      const int* __restrict__ et,
                                                      float* __restrict__ agg) {
  int idx = blockIdx.x * blockDim.x + threadIdx.x;   // over E * C
  if (idx >= Ee * Cx) return;
  int e = idx / Cx;
  int c = idx - e * Cx;
  int src = ei[e];
  int dst = ei[Ee + e];
  int r   = et[e];
  size_t segOff = ((size_t)dst * Rr + r) * Cx + c;
  size_t srcOff = (size_t)src * Cx + c;
#pragma unroll
  for (int b = 0; b < Bx; ++b) {
    atomicAdd(&agg[(size_t)b * ((size_t)Lx * Rr * Cx) + segOff],
              v[(size_t)b * ((size_t)Lx * Cx) + srcOff]);
  }
}

// ---------------- depthwise conv KSxKS, channel-last layout ----------------
template <int KS>
__global__ __launch_bounds__(256) void dwconv_kernel(const float* __restrict__ v,
                                                     const float* __restrict__ w,
                                                     const float* __restrict__ bias,
                                                     float* __restrict__ out) {
  int idx = blockIdx.x * blockDim.x + threadIdx.x;   // over B*H*W*C
  if (idx >= (int)(M * Cx)) return;
  int c  = idx % Cx;
  int t  = idx / Cx;           // b*H*W + y*W + x
  int xw = t % Ww;
  int t2 = t / Ww;
  int y  = t2 % Hh;
  int b  = t2 / Hh;
  constexpr int P = KS / 2;
  float s = 0.0f;
#pragma unroll
  for (int ky = 0; ky < KS; ++ky) {
    int yy = y + ky - P;
    if (yy < 0 || yy >= Hh) continue;
#pragma unroll
    for (int kx = 0; kx < KS; ++kx) {
      int xx = xw + kx - P;
      if (xx < 0 || xx >= Ww) continue;
      s += v[((size_t)b * Lx + (size_t)yy * Ww + xx) * Cx + c] *
           w[c * KS * KS + ky * KS + kx];
    }
  }
  out[idx] = s + bias[c];
}

// ---------------- gate + mean + pack upd[M, RT*C] as f16 ----------------
__global__ __launch_bounds__(256) void pack_kernel(const _Float16* __restrict__ h16,
                                                   const float* __restrict__ gw,
                                                   const float* __restrict__ gb,
                                                   const float* __restrict__ agg,
                                                   const float* __restrict__ cnt,
                                                   const float* __restrict__ c3,
                                                   const float* __restrict__ c5,
                                                   _Float16* __restrict__ upd) {
  int m = (blockIdx.x << 3) + (threadIdx.x >> 5);
  if (m >= (int)M) return;
  int lane = threadIdx.x & 31;
  int b = m / Lx;
  int l = m - b * Lx;

  // gate = sigmoid(h @ gate_w + gate_b), 5 relations, reduced across the wave
  float g[RTt] = {0.f, 0.f, 0.f, 0.f, 0.f};
#pragma unroll
  for (int j = 0; j < 3; ++j) {
    int c = lane + (j << 5);
    float hv = (float)h16[(size_t)m * Cx + c];
#pragma unroll
    for (int r = 0; r < RTt; ++r) g[r] += hv * gw[c * RTt + r];
  }
#pragma unroll
  for (int r = 0; r < RTt; ++r) {
    float s = g[r];
#pragma unroll
    for (int mm = 16; mm > 0; mm >>= 1) s += __shfl_xor(s, mm, 32);
    g[r] = 1.0f / (1.0f + expf(-(s + gb[r])));
  }

  size_t ub = (size_t)m * (RTt * Cx);
#pragma unroll
  for (int j = 0; j < 3; ++j) {
    int c = lane + (j << 5);
#pragma unroll
    for (int r = 0; r < Rr; ++r) {
      float cv = cnt[l * Rr + r];
      cv = cv < 1.0f ? 1.0f : cv;
      float a = agg[((size_t)b * Lx * Rr + (size_t)l * Rr + r) * Cx + c] / cv;
      upd[ub + r * Cx + c] = (_Float16)(a * g[r]);
    }
    upd[ub + 3 * Cx + c] = (_Float16)(c3[(size_t)m * Cx + c] * g[3]);
    upd[ub + 4 * Cx + c] = (_Float16)(c5[(size_t)m * Cx + c] * g[4]);
  }
}

// ---------------- host side ----------------
static inline size_t align256(size_t x) { return (x + 255) & ~(size_t)255; }

extern "C" void kernel_launch(void* const* d_in, const int* in_sizes, int n_in,
                              void* d_out, int out_size, void* d_ws, size_t ws_size,
                              hipStream_t stream) {
  const float* x        = (const float*)d_in[0];
  const int*   ei       = (const int*)d_in[1];
  const int*   et       = (const int*)d_in[2];
  const float* norm1_g  = (const float*)d_in[5];
  const float* norm1_b  = (const float*)d_in[6];
  const float* value_w  = (const float*)d_in[7];
  const float* value_b  = (const float*)d_in[8];
  const float* gate_w   = (const float*)d_in[9];
  const float* gate_b   = (const float*)d_in[10];
  const float* ctx_k3   = (const float*)d_in[11];
  const float* ctx_b3   = (const float*)d_in[12];
  const float* ctx_k5   = (const float*)d_in[13];
  const float* ctx_b5   = (const float*)d_in[14];
  const float* rel_w    = (const float*)d_in[15];
  const float* rel_b    = (const float*)d_in[16];
  const float* proj_w   = (const float*)d_in[17];
  const float* proj_b   = (const float*)d_in[18];
  const float* norm2_g  = (const float*)d_in[19];
  const float* norm2_b  = (const float*)d_in[20];
  const float* fc1_w    = (const float*)d_in[21];
  const float* fc1_b    = (const float*)d_in[22];
  const float* fc2_w    = (const float*)d_in[23];
  const float* fc2_b    = (const float*)d_in[24];
  float* out = (float*)d_out;

  // ---- workspace arena (lifetime-aliased) ----
  char* ws = (char*)d_ws;
  size_t o = 0;
  size_t o_h16 = o; o += align256(M * Cx * sizeof(_Float16));                 // LN1 out (f16)
  size_t o_v32 = o; o += align256(M * Cx * sizeof(float));                    // value out (f32)
  size_t o_cnt = o; o += align256((size_t)Lx * Rr * sizeof(float));          // segment counts
  size_t o_agg = o; o += align256((size_t)Bx * Lx * Rr * Cx * sizeof(float)); // scatter acc
  size_t o_c3  = o; o += align256(M * Cx * sizeof(float));                    // 3x3 ctx
  size_t o_c5  = o; o += align256(M * Cx * sizeof(float));                    // 5x5 ctx
  size_t o_upd = o; o += align256(M * RTt * Cx * sizeof(_Float16));           // packed upd (f16)
  size_t o_wv  = o; o += align256((size_t)Cx * Cx * sizeof(_Float16));        // value_w^T f16
  size_t o_wr  = o; o += align256((size_t)Cx * RTt * Cx * sizeof(_Float16));  // rel_w^T f16
  size_t o_wp  = o; o += align256((size_t)Cx * Cx * sizeof(_Float16));        // proj_w^T f16
  size_t o_w1  = o; o += align256((size_t)FFN * Cx * sizeof(_Float16));       // fc1_w^T f16
  size_t o_w2  = o; o += align256((size_t)Cx * FFN * sizeof(_Float16));       // fc2_w^T f16
  // aliases (dead-after-use regions)
  size_t o_gmid = o_agg;   // rel-mix gelu output (f16), agg dead after pack
  size_t o_x1   = o_c3;    // residual-1 (f32), c3 dead after pack
  size_t o_h2   = o_c5;    // LN2 out (f16), c5 dead after pack
  size_t o_t    = o_upd;   // FFN mid (f16), upd dead after rel gemm

  _Float16* h16  = (_Float16*)(ws + o_h16);
  float*    v32  = (float*)(ws + o_v32);
  float*    cnt  = (float*)(ws + o_cnt);
  float*    agg  = (float*)(ws + o_agg);
  float*    c3   = (float*)(ws + o_c3);
  float*    c5   = (float*)(ws + o_c5);
  _Float16* upd  = (_Float16*)(ws + o_upd);
  _Float16* wTv  = (_Float16*)(ws + o_wv);
  _Float16* wTr  = (_Float16*)(ws + o_wr);
  _Float16* wTp  = (_Float16*)(ws + o_wp);
  _Float16* wT1  = (_Float16*)(ws + o_w1);
  _Float16* wT2  = (_Float16*)(ws + o_w2);
  _Float16* gmid = (_Float16*)(ws + o_gmid);
  float*    x1   = (float*)(ws + o_x1);
  _Float16* h2   = (_Float16*)(ws + o_h2);
  _Float16* t16  = (_Float16*)(ws + o_t);

  const int TPB = 256;
  auto blks = [](size_t n, int tpb) { return (int)((n + tpb - 1) / tpb); };

  // 1) weight conversion f32 [K,N] -> f16 [N,K]
  transpose_f16_kernel<<<blks((size_t)Cx * Cx, TPB), TPB, 0, stream>>>(value_w, wTv, Cx, Cx);
  transpose_f16_kernel<<<blks((size_t)RTt * Cx * Cx, TPB), TPB, 0, stream>>>(rel_w, wTr, RTt * Cx, Cx);
  transpose_f16_kernel<<<blks((size_t)Cx * Cx, TPB), TPB, 0, stream>>>(proj_w, wTp, Cx, Cx);
  transpose_f16_kernel<<<blks((size_t)Cx * FFN, TPB), TPB, 0, stream>>>(fc1_w, wT1, Cx, FFN);
  transpose_f16_kernel<<<blks((size_t)FFN * Cx, TPB), TPB, 0, stream>>>(fc2_w, wT2, FFN, Cx);

  // 2) LN1
  ln_kernel<<<blks(M, 8), TPB, 0, stream>>>(x, norm1_g, norm1_b, h16);

  // 3) v = h @ value_w + value_b   (WMMA, f32 out)
  {
    int tiles = (int)(M / 16) * (Cx / 16);
    wmma_gemm<false, false, true, false><<<blks(tiles, 8), TPB, 0, stream>>>(
        h16, wTv, value_b, nullptr, v32, nullptr, (int)(M / 16), Cx / 16, Cx);
  }

  // 4) segment counts + scatter-add
  hipMemsetAsync(cnt, 0, (size_t)Lx * Rr * sizeof(float), stream);
  hipMemsetAsync(agg, 0, (size_t)Bx * Lx * Rr * Cx * sizeof(float), stream);
  count_kernel<<<blks(Ee, TPB), TPB, 0, stream>>>(ei, et, cnt);
  scatter_kernel<<<blks((size_t)Ee * Cx, TPB), TPB, 0, stream>>>(v32, ei, et, agg);

  // 5) depthwise context convs
  dwconv_kernel<3><<<blks(M * Cx, TPB), TPB, 0, stream>>>(v32, ctx_k3, ctx_b3, c3);
  dwconv_kernel<5><<<blks(M * Cx, TPB), TPB, 0, stream>>>(v32, ctx_k5, ctx_b5, c5);

  // 6) gate + mean + pack upd (f16, [M, RT*C])
  pack_kernel<<<blks(M, 8), TPB, 0, stream>>>(h16, gate_w, gate_b, agg, cnt, c3, c5, upd);

  // 7) gmid = gelu(upd @ rel_w + rel_b)   (WMMA K=480, f16 out)
  {
    int tiles = (int)(M / 16) * (Cx / 16);
    wmma_gemm<true, false, false, true><<<blks(tiles, 8), TPB, 0, stream>>>(
        upd, wTr, rel_b, nullptr, nullptr, gmid, (int)(M / 16), Cx / 16, RTt * Cx);
  }

  // 8) x1 = x + gmid @ proj_w + proj_b   (WMMA + residual, f32 out)
  {
    int tiles = (int)(M / 16) * (Cx / 16);
    wmma_gemm<false, true, true, false><<<blks(tiles, 8), TPB, 0, stream>>>(
        gmid, wTp, proj_b, x, x1, nullptr, (int)(M / 16), Cx / 16, Cx);
  }

  // 9) LN2
  ln_kernel<<<blks(M, 8), TPB, 0, stream>>>(x1, norm2_g, norm2_b, h2);

  // 10) t = gelu(h2 @ fc1_w + fc1_b)   (WMMA N=384, f16 out)
  {
    int tiles = (int)(M / 16) * (FFN / 16);
    wmma_gemm<true, false, false, true><<<blks(tiles, 8), TPB, 0, stream>>>(
        h2, wT1, fc1_b, nullptr, nullptr, t16, (int)(M / 16), FFN / 16, Cx);
  }

  // 11) out = x1 + t @ fc2_w + fc2_b   (WMMA K=384 + residual, f32 out)
  {
    int tiles = (int)(M / 16) * (Cx / 16);
    wmma_gemm<false, true, true, false><<<blks(tiles, 8), TPB, 0, stream>>>(
        t16, wT2, fc2_b, x1, out, nullptr, (int)(M / 16), Cx / 16, FFN);
  }
}
